// TransformerBlock_54941221651188
// MI455X (gfx1250) — matrix operand order repaired and checked by reading, not compiled, not run on hardware
//
#include <hip/hip_runtime.h>
#include <hip/hip_bf16.h>
#include <math.h>

typedef __attribute__((ext_vector_type(16))) __bf16 v16bf;
typedef __attribute__((ext_vector_type(8)))  __bf16 v8bf;
typedef __attribute__((ext_vector_type(8)))  float  v8f;

#define EMB   1024
#define HEADS 16
#define HDIM  64
#define FFDIM 4096
#define SEQ   2048
#define BATCH 2
#define TOK   (BATCH * SEQ)

#define BM 128
#define BN 128
#define BK 32

// ---- CDNA5 async global->LDS DMA (ASYNCcnt-tracked), guarded so we always compile
#if defined(__has_builtin)
#  if __has_builtin(__builtin_amdgcn_global_load_async_to_lds_b128) && \
      __has_builtin(__builtin_amdgcn_s_wait_asynccnt)
#    define HAVE_ASYNC_LDS 1
#  endif
#endif
#ifndef HAVE_ASYNC_LDS
#  define HAVE_ASYNC_LDS 0
#endif

#if HAVE_ASYNC_LDS
typedef int v4i_vs __attribute__((vector_size(16)));
__device__ __forceinline__ void async_copy16(const void* g, void* l) {
  auto gp = (__attribute__((address_space(1))) v4i_vs*)(unsigned long long)g;
  auto lp = (__attribute__((address_space(3))) v4i_vs*)(unsigned)(unsigned long long)l;
  __builtin_amdgcn_global_load_async_to_lds_b128(gp, lp, 0, 0);
}
__device__ __forceinline__ void async_wait0() { __builtin_amdgcn_s_wait_asynccnt(0); }
#endif

__device__ __forceinline__ v16bf mk16(v8bf lo, v8bf hi) {
  v16bf r;
#pragma unroll
  for (int i = 0; i < 8; ++i) { r[i] = lo[i]; r[i + 8] = hi[i]; }
  return r;
}

__device__ __forceinline__ float gelu_fn(float x) {
  const float c = 0.7978845608028654f;  // sqrt(2/pi)
  return 0.5f * x * (1.0f + tanhf(c * (x + 0.044715f * x * x * x)));
}

// ---------------------------------------------------------------- cast f32->bf16
__global__ void cast_f32_to_bf16(const float* __restrict__ in,
                                 __bf16* __restrict__ out, long n) {
  long i = (long)blockIdx.x * blockDim.x + threadIdx.x;
  long stride = (long)gridDim.x * blockDim.x;
  for (; i < n; i += stride) out[i] = (__bf16)in[i];
}

// ---------------------------------------------------------------- LayerNorm + cast
__global__ __launch_bounds__(256) void ln_cast_kernel(
    const float* __restrict__ x, const float* __restrict__ scale,
    const float* __restrict__ shift, __bf16* __restrict__ out) {
  __shared__ float s1[256];
  __shared__ float s2[256];
  const int row = blockIdx.x;
  const int t = threadIdx.x;
  const float4 v = *(const float4*)(x + (size_t)row * EMB + t * 4);
  s1[t] = v.x + v.y + v.z + v.w;
  s2[t] = v.x * v.x + v.y * v.y + v.z * v.z + v.w * v.w;
  __syncthreads();
  for (int off = 128; off > 0; off >>= 1) {
    if (t < off) { s1[t] += s1[t + off]; s2[t] += s2[t + off]; }
    __syncthreads();
  }
  const float mean = s1[0] * (1.0f / EMB);
  const float var = s2[0] * (1.0f / EMB) - mean * mean;
  const float rstd = rsqrtf(var + 1e-5f);
  const float4 sc = *(const float4*)(scale + t * 4);
  const float4 sh = *(const float4*)(shift + t * 4);
  __bf16* o = out + (size_t)row * EMB + t * 4;
  o[0] = (__bf16)(sc.x * ((v.x - mean) * rstd) + sh.x);
  o[1] = (__bf16)(sc.y * ((v.y - mean) * rstd) + sh.y);
  o[2] = (__bf16)(sc.z * ((v.z - mean) * rstd) + sh.z);
  o[3] = (__bf16)(sc.w * ((v.w - mean) * rstd) + sh.w);
}

// ---------------------------------------------------------------- bf16 WMMA GEMM
// C[M,N] = A[M,K] * B[K,N]  (+bias, optional GELU, +fp32 residual)
// 256 threads = 8 waves; wave grid 2(M) x 4(N); each wave 64x32 = 8 WMMA tiles.
// Double-buffered LDS; next tile is in flight (async DMA or reg-buffered global
// loads) while the current tile runs through v_wmma.
__global__ __launch_bounds__(256) void gemm_bf16_wmma(
    const __bf16* __restrict__ A, const __bf16* __restrict__ B,
    const float* __restrict__ bias, const float* __restrict__ resid,
    float* __restrict__ outF, __bf16* __restrict__ outH,
    int M, int N, int K, int doGelu) {
  __shared__ __attribute__((aligned(16))) __bf16 lsA[2][BM * BK];  // [m][k]
  __shared__ __attribute__((aligned(16))) __bf16 lsB[2][BN * BK];  // [n][k]

  const int tid = threadIdx.x;
  const int lane = tid & 31;
  const int wave = tid >> 5;
  const int waveM = wave & 1;
  const int waveN = wave >> 1;
  const int hf = lane >> 4;
  const int l16 = lane & 15;
  const int m0 = blockIdx.y * BM;
  const int n0 = blockIdx.x * BN;

  int mA[2], kA[2], kB[2], nB[2];
#pragma unroll
  for (int c = 0; c < 2; ++c) {
    int l = tid * 16 + c * 8;
    mA[c] = l >> 5; kA[c] = l & 31;   // A tile: 128x32
    kB[c] = l >> 7; nB[c] = l & 127;  // B tile: 32x128
  }

  v8f acc[4][2];
#pragma unroll
  for (int i = 0; i < 4; ++i)
#pragma unroll
    for (int j = 0; j < 2; ++j)
#pragma unroll
      for (int r = 0; r < 8; ++r) acc[i][j][r] = 0.0f;

  uint4 aR[2];
  v8bf bR[2];

  // ---- prologue: stage K-tile 0 into buffer 0
#pragma unroll
  for (int c = 0; c < 2; ++c) {
#if HAVE_ASYNC_LDS
    async_copy16(&A[(size_t)(m0 + mA[c]) * K + kA[c]], &lsA[0][mA[c] * BK + kA[c]]);
#else
    aR[c] = *(const uint4*)&A[(size_t)(m0 + mA[c]) * K + kA[c]];
#endif
    bR[c] = *(const v8bf*)&B[(size_t)kB[c] * N + n0 + nB[c]];
  }
#if !HAVE_ASYNC_LDS
#pragma unroll
  for (int c = 0; c < 2; ++c) *(uint4*)&lsA[0][mA[c] * BK + kA[c]] = aR[c];
#endif
#pragma unroll
  for (int c = 0; c < 2; ++c)
#pragma unroll
    for (int j = 0; j < 8; ++j) lsB[0][(nB[c] + j) * BK + kB[c]] = bR[c][j];
#if HAVE_ASYNC_LDS
  async_wait0();
#endif
  __syncthreads();

  const int nk = K / BK;
  for (int t = 0; t < nk; ++t) {
    const int cur = t & 1;
    const int nxt = cur ^ 1;
    const int k1 = (t + 1) * BK;
    const bool more = (t + 1) < nk;

    if (more) {  // launch next tile's traffic before any math
#pragma unroll
      for (int c = 0; c < 2; ++c) {
#if HAVE_ASYNC_LDS
        async_copy16(&A[(size_t)(m0 + mA[c]) * K + k1 + kA[c]],
                     &lsA[nxt][mA[c] * BK + kA[c]]);
#else
        aR[c] = *(const uint4*)&A[(size_t)(m0 + mA[c]) * K + k1 + kA[c]];
#endif
        bR[c] = *(const v8bf*)&B[(size_t)(k1 + kB[c]) * N + n0 + nB[c]];
      }
      if (k1 + BK < K) {  // global_prefetch_b8 one tile further out
        __builtin_prefetch(&A[(size_t)(m0 + (tid >> 1)) * K + k1 + BK], 0, 2);
        __builtin_prefetch(&B[(size_t)(k1 + BK + (tid & 31)) * N + n0], 0, 2);
      }
    }

    // ---- compute on current buffer
    v16bf af[4], bfr[2];
#pragma unroll
    for (int tm = 0; tm < 4; ++tm) {
      int row = waveM * 64 + tm * 16 + l16;
      int base = row * BK + hf * 8;  // lane<16: K0-7/16-23, lane>=16: K8-15/24-31
      af[tm] = mk16(*(const v8bf*)&lsA[cur][base], *(const v8bf*)&lsA[cur][base + 16]);
    }
#pragma unroll
    for (int tn = 0; tn < 2; ++tn) {
      int col = waveN * 32 + tn * 16 + l16;
      int base = col * BK + hf * 8;  // lane<16: K0-7/16-23, lane>=16: K8-15/24-31
      bfr[tn] = mk16(*(const v8bf*)&lsB[cur][base], *(const v8bf*)&lsB[cur][base + 16]);
    }
#pragma unroll
    for (int tm = 0; tm < 4; ++tm)
#pragma unroll
      for (int tn = 0; tn < 2; ++tn)
        acc[tm][tn] = __builtin_amdgcn_wmma_f32_16x16x32_bf16(
            false, af[tm], false, bfr[tn], (short)0, acc[tm][tn], false, false);

    // ---- drain next tile into LDS
    if (more) {
#if !HAVE_ASYNC_LDS
#pragma unroll
      for (int c = 0; c < 2; ++c) *(uint4*)&lsA[nxt][mA[c] * BK + kA[c]] = aR[c];
#endif
#pragma unroll
      for (int c = 0; c < 2; ++c)
#pragma unroll
        for (int j = 0; j < 8; ++j) lsB[nxt][(nB[c] + j) * BK + kB[c]] = bR[c][j];
#if HAVE_ASYNC_LDS
      async_wait0();
#endif
    }
    __syncthreads();
  }

  // ---- epilogue (C layout: VGPR r -> row r (lanes 0-15) / r+8 (lanes 16-31))
#pragma unroll
  for (int tm = 0; tm < 4; ++tm)
#pragma unroll
    for (int tn = 0; tn < 2; ++tn)
#pragma unroll
      for (int r = 0; r < 8; ++r) {
        int m = m0 + waveM * 64 + tm * 16 + r + hf * 8;
        int n = n0 + waveN * 32 + tn * 16 + l16;
        float val = acc[tm][tn][r];
        if (bias) val += bias[n];
        if (doGelu) val = gelu_fn(val);
        if (resid) val += resid[(size_t)m * N + n];
        if (outF) outF[(size_t)m * N + n] = val;
        if (outH) outH[(size_t)m * N + n] = (__bf16)val;
      }
}

// ---------------------------------------------------------------- flash attention
// block = 128 threads (4 waves); each wave: 16 query rows x full head (64).
// grid = (SEQ/64, HEADS, BATCH). Streams keys/values in double-buffered blocks
// of 32; next block's K goes via async DMA (V needs an element transpose so it
// stays reg->scatter).
__global__ __launch_bounds__(128) void attn_kernel(
    const __bf16* __restrict__ q, const __bf16* __restrict__ k,
    const __bf16* __restrict__ v, __bf16* __restrict__ ctx) {
  __shared__ __attribute__((aligned(16))) __bf16 lsK[2][32 * HDIM];   // [key][d]
  __shared__ __attribute__((aligned(16))) __bf16 lsVt[2][HDIM * 32];  // [d][key]
  __shared__ __attribute__((aligned(16))) __bf16 lsP[4][16 * 32];     // per-wave P

  const int tid = threadIdx.x;
  const int lane = tid & 31;
  const int wave = tid >> 5;
  const int hf = lane >> 4;
  const int l16 = lane & 15;
  const int h = blockIdx.y;
  const int b = blockIdx.z;
  const size_t tokBase = (size_t)b * SEQ;
  const int qRow0 = blockIdx.x * 64 + wave * 16;
  const float scale = 0.125f;  // 1/sqrt(HDIM)

  int keyI[2], dI[2];
#pragma unroll
  for (int c = 0; c < 2; ++c) {
    int l = tid * 16 + c * 8;
    keyI[c] = l >> 6; dI[c] = l & 63;
  }

  // Q fragments (16 rows x K=64 -> two K=32 A-frags), reused for all key blocks
  v16bf qa[2];
#pragma unroll
  for (int f = 0; f < 2; ++f) {
    const __bf16* qp = q + (tokBase + qRow0 + l16) * EMB + h * HDIM + f * 32 + hf * 8;
    qa[f] = mk16(*(const v8bf*)qp, *(const v8bf*)(qp + 16));
  }

  v8f accv[4];
#pragma unroll
  for (int tn = 0; tn < 4; ++tn)
#pragma unroll
    for (int r = 0; r < 8; ++r) accv[tn][r] = 0.0f;
  float mrun[8], lrun[8];
#pragma unroll
  for (int r = 0; r < 8; ++r) { mrun[r] = -__builtin_inff(); lrun[r] = 0.0f; }

  uint4 kR[2];
  v8bf vR[2];

  // ---- prologue: stage key-block 0 into buffer 0
#pragma unroll
  for (int c = 0; c < 2; ++c) {
    const size_t gix = (tokBase + keyI[c]) * EMB + h * HDIM + dI[c];
#if HAVE_ASYNC_LDS
    async_copy16(&k[gix], &lsK[0][keyI[c] * HDIM + dI[c]]);
#else
    kR[c] = *(const uint4*)&k[gix];
#endif
    vR[c] = *(const v8bf*)&v[gix];
  }
#if !HAVE_ASYNC_LDS
#pragma unroll
  for (int c = 0; c < 2; ++c) *(uint4*)&lsK[0][keyI[c] * HDIM + dI[c]] = kR[c];
#endif
#pragma unroll
  for (int c = 0; c < 2; ++c)
#pragma unroll
    for (int j = 0; j < 8; ++j) lsVt[0][(dI[c] + j) * 32 + keyI[c]] = vR[c][j];
#if HAVE_ASYNC_LDS
  async_wait0();
#endif
  __syncthreads();

  for (int kb = 0; kb < SEQ; kb += 32) {
    const int cur = (kb >> 5) & 1;
    const int nxt = cur ^ 1;
    const bool more = (kb + 32) < SEQ;

    if (more) {  // launch next block's traffic before the math
#pragma unroll
      for (int c = 0; c < 2; ++c) {
        const size_t gix = (tokBase + kb + 32 + keyI[c]) * EMB + h * HDIM + dI[c];
#if HAVE_ASYNC_LDS
        async_copy16(&k[gix], &lsK[nxt][keyI[c] * HDIM + dI[c]]);
#else
        kR[c] = *(const uint4*)&k[gix];
#endif
        vR[c] = *(const v8bf*)&v[gix];
      }
    }

    // scores: S(16x32) = Q(16x64) x K_blk^T -> two 16x16 tiles, 4 WMMAs
    v8f c0, c1;
#pragma unroll
    for (int r = 0; r < 8; ++r) { c0[r] = 0.0f; c1[r] = 0.0f; }
#pragma unroll
    for (int f = 0; f < 2; ++f) {
      int ba = l16 * HDIM + f * 32 + hf * 8;
      v16bf b0 = mk16(*(const v8bf*)&lsK[cur][ba], *(const v8bf*)&lsK[cur][ba + 16]);
      int bb = (16 + l16) * HDIM + f * 32 + hf * 8;
      v16bf b1 = mk16(*(const v8bf*)&lsK[cur][bb], *(const v8bf*)&lsK[cur][bb + 16]);
      c0 = __builtin_amdgcn_wmma_f32_16x16x32_bf16(false, qa[f], false, b0, (short)0, c0, false, false);
      c1 = __builtin_amdgcn_wmma_f32_16x16x32_bf16(false, qa[f], false, b1, (short)0, c1, false, false);
    }

    // online softmax; rows live in lane groups of 16 (width-16 butterflies)
#pragma unroll
    for (int r = 0; r < 8; ++r) {
      float a0 = c0[r] * scale, a1 = c1[r] * scale;
      float rm = fmaxf(a0, a1);
#pragma unroll
      for (int mseq = 8; mseq >= 1; mseq >>= 1) rm = fmaxf(rm, __shfl_xor(rm, mseq, 16));
      float mnew = fmaxf(mrun[r], rm);
      float alpha = __expf(mrun[r] - mnew);
      float p0 = __expf(a0 - mnew), p1 = __expf(a1 - mnew);
      float ps = p0 + p1;
#pragma unroll
      for (int mseq = 8; mseq >= 1; mseq >>= 1) ps += __shfl_xor(ps, mseq, 16);
      lrun[r] = lrun[r] * alpha + ps;
      mrun[r] = mnew;
#pragma unroll
      for (int tn = 0; tn < 4; ++tn) accv[tn][r] *= alpha;
      int row = r + hf * 8;
      lsP[wave][row * 32 + l16] = (__bf16)p0;
      lsP[wave][row * 32 + 16 + l16] = (__bf16)p1;
    }

    // ctx += P(16x32) x V_blk(32x64): 4 WMMAs (per-wave LDS, DS is in-order per wave)
    {
      int base = l16 * 32 + hf * 8;
      v16bf pa = mk16(*(const v8bf*)&lsP[wave][base], *(const v8bf*)&lsP[wave][base + 16]);
#pragma unroll
      for (int tn = 0; tn < 4; ++tn) {
        int col = tn * 16 + l16;
        v16bf vb = mk16(*(const v8bf*)&lsVt[cur][col * 32 + hf * 8],
                        *(const v8bf*)&lsVt[cur][col * 32 + hf * 8 + 16]);
        accv[tn] = __builtin_amdgcn_wmma_f32_16x16x32_bf16(
            false, pa, false, vb, (short)0, accv[tn], false, false);
      }
    }

    // drain next block into LDS
    if (more) {
#if !HAVE_ASYNC_LDS
#pragma unroll
      for (int c = 0; c < 2; ++c) *(uint4*)&lsK[nxt][keyI[c] * HDIM + dI[c]] = kR[c];
#endif
#pragma unroll
      for (int c = 0; c < 2; ++c)
#pragma unroll
        for (int j = 0; j < 8; ++j) lsVt[nxt][(dI[c] + j) * 32 + keyI[c]] = vR[c][j];
#if HAVE_ASYNC_LDS
      async_wait0();
#endif
    }
    __syncthreads();
  }

#pragma unroll
  for (int tn = 0; tn < 4; ++tn)
#pragma unroll
    for (int r = 0; r < 8; ++r) {
      int row = qRow0 + r + hf * 8;
      int n = tn * 16 + l16;
      ctx[(tokBase + row) * EMB + h * HDIM + n] = (__bf16)(accv[tn][r] / lrun[r]);
    }
}

// ---------------------------------------------------------------- host launcher
extern "C" void kernel_launch(void* const* d_in, const int* in_sizes, int n_in,
                              void* d_out, int out_size, void* d_ws, size_t ws_size,
                              hipStream_t stream) {
  const float* x    = (const float*)d_in[0];
  const float* wq   = (const float*)d_in[1];
  const float* wk   = (const float*)d_in[2];
  const float* wv   = (const float*)d_in[3];
  const float* wo   = (const float*)d_in[4];
  const float* b_o  = (const float*)d_in[5];
  const float* wff1 = (const float*)d_in[6];
  const float* bff1 = (const float*)d_in[7];
  const float* wff2 = (const float*)d_in[8];
  const float* bff2 = (const float*)d_in[9];
  const float* ln1s = (const float*)d_in[10];
  const float* ln1b = (const float*)d_in[11];
  const float* ln2s = (const float*)d_in[12];
  const float* ln2b = (const float*)d_in[13];
  float* out = (float*)d_out;

  char* p = (char*)d_ws;
  auto take = [&](size_t bytes) { void* r = (void*)p; p += bytes; return r; };
  __bf16* wq_h   = (__bf16*)take((size_t)EMB * EMB * 2);
  __bf16* wk_h   = (__bf16*)take((size_t)EMB * EMB * 2);
  __bf16* wv_h   = (__bf16*)take((size_t)EMB * EMB * 2);
  __bf16* wo_h   = (__bf16*)take((size_t)EMB * EMB * 2);
  __bf16* wff1_h = (__bf16*)take((size_t)EMB * FFDIM * 2);
  __bf16* wff2_h = (__bf16*)take((size_t)FFDIM * EMB * 2);
  __bf16* h1     = (__bf16*)take((size_t)TOK * EMB * 2);
  __bf16* qb     = (__bf16*)take((size_t)TOK * EMB * 2);
  __bf16* kb     = (__bf16*)take((size_t)TOK * EMB * 2);
  __bf16* vb     = (__bf16*)take((size_t)TOK * EMB * 2);
  __bf16* ctxb   = (__bf16*)take((size_t)TOK * EMB * 2);
  float*  x1     = (float*) take((size_t)TOK * EMB * 4);
  __bf16* h2     = (__bf16*)take((size_t)TOK * EMB * 2);
  __bf16* ff1o   = (__bf16*)take((size_t)TOK * FFDIM * 2);

  // weight casts (bf16 weights stay hot in the 192MB L2)
  cast_f32_to_bf16<<<1024, 256, 0, stream>>>(wq,   wq_h,   (long)EMB * EMB);
  cast_f32_to_bf16<<<1024, 256, 0, stream>>>(wk,   wk_h,   (long)EMB * EMB);
  cast_f32_to_bf16<<<1024, 256, 0, stream>>>(wv,   wv_h,   (long)EMB * EMB);
  cast_f32_to_bf16<<<1024, 256, 0, stream>>>(wo,   wo_h,   (long)EMB * EMB);
  cast_f32_to_bf16<<<2048, 256, 0, stream>>>(wff1, wff1_h, (long)EMB * FFDIM);
  cast_f32_to_bf16<<<2048, 256, 0, stream>>>(wff2, wff2_h, (long)FFDIM * EMB);

  // LN1 -> h1 (bf16)
  ln_cast_kernel<<<TOK, 256, 0, stream>>>(x, ln1s, ln1b, h1);

  // QKV projections
  dim3 gE(EMB / BN, TOK / BM);
  gemm_bf16_wmma<<<gE, 256, 0, stream>>>(h1, wq_h, nullptr, nullptr, nullptr, qb, TOK, EMB, EMB, 0);
  gemm_bf16_wmma<<<gE, 256, 0, stream>>>(h1, wk_h, nullptr, nullptr, nullptr, kb, TOK, EMB, EMB, 0);
  gemm_bf16_wmma<<<gE, 256, 0, stream>>>(h1, wv_h, nullptr, nullptr, nullptr, vb, TOK, EMB, EMB, 0);

  // attention -> ctx (bf16)
  attn_kernel<<<dim3(SEQ / 64, HEADS, BATCH), 128, 0, stream>>>(qb, kb, vb, ctxb);

  // O projection + b_o + residual(x) -> x1 (fp32)
  gemm_bf16_wmma<<<gE, 256, 0, stream>>>(ctxb, wo_h, b_o, x, x1, nullptr, TOK, EMB, EMB, 0);

  // LN2 -> h2 (bf16)
  ln_cast_kernel<<<TOK, 256, 0, stream>>>(x1, ln2s, ln2b, h2);

  // FF1 + bias + GELU -> ff1o (bf16)
  dim3 gF(FFDIM / BN, TOK / BM);
  gemm_bf16_wmma<<<gF, 256, 0, stream>>>(h2, wff1_h, bff1, nullptr, nullptr, ff1o, TOK, FFDIM, EMB, 1);

  // FF2 + bias + residual(x1) -> out (fp32)
  gemm_bf16_wmma<<<gE, 256, 0, stream>>>(ff1o, wff2_h, bff2, x1, out, nullptr, TOK, EMB, FFDIM, 0);
}
